// DeepRGCNEncoder_592705487137
// MI455X (gfx1250) — matrix operand order repaired and checked by reading, not compiled, not run on hardware
//
#include <hip/hip_runtime.h>
#include <hip/hip_fp16.h>

// Problem constants (from reference)
#define NNODES 10000
#define NEDGES 80000
#define ESL    (NEDGES + NNODES)   // edges + self loops
#define HID    500
#define NHEADS 4
#define NRELS  32
#define NBASES 30
#define NGCN   3

typedef __attribute__((ext_vector_type(16))) _Float16 v16h;
typedef __attribute__((ext_vector_type(8)))  float    v8f;

union HFrag { uint4 u[2]; v16h h; };

// ---------------------------------------------------------------------------
// float atomic max via CAS (finite values only)
// ---------------------------------------------------------------------------
__device__ __forceinline__ void atomic_max_f32(float* addr, float val) {
    int* ia = reinterpret_cast<int*>(addr);
    int cur = __float_as_int(*addr);
    while (__int_as_float(cur) < val) {
        int old = atomicCAS(ia, cur, __float_as_int(val));
        if (old == cur) break;
        cur = old;
    }
}

// ---------------------------------------------------------------------------
// Generic WMMA GEMM: C[M,N] = (beta1 ? C : 0) + A[M,K] @ B[K,N]
// fp32 in/out, f16 compute via v_wmma_f32_16x16x32_f16 (wave32).
// Block: 256 threads = 8 waves; tile 64x128; K-step 32.
// A panel staged fp32 in LDS via GLOBAL_LOAD_ASYNC_TO_LDS_B128 (byte copy,
// ASYNCcnt), converted to f16 at fragment-load time; B panel staged f16
// transposed (Bs[n][k]) so fragments are two contiguous 16B ds reads.
// ---------------------------------------------------------------------------
#define TM 64
#define TN 128
#define TK 32

__global__ __launch_bounds__(256) void wmma_gemm_f16f32(
    const float* __restrict__ A, const float* __restrict__ B,
    float* __restrict__ C, int M, int N, int K, int beta1)
{
    __shared__ float    As32[TM][TK];   // fp32 [m][k], k contiguous (8 KB)
    __shared__ _Float16 Bs[TN][TK];     // f16 transposed [n][k]      (8 KB)

    const int tid  = threadIdx.x;
    const int wave = tid >> 5;          // 0..7
    const int lane = tid & 31;
    const int bm   = blockIdx.y * TM;
    const int bn   = blockIdx.x * TN;

    // wave -> (row tile, 4 col tiles): tile grid is 4 rows x 8 cols of 16x16
    const int wm  = (wave & 3) * 16;
    const int wn0 = (wave >> 2) * 64;

    v8f acc[4] = {v8f{}, v8f{}, v8f{}, v8f{}};

    const bool kAligned = ((K & 3) == 0);          // 16B-aligned rows + chunks
    const unsigned asBase = (unsigned)(size_t)&As32[0][0];

    const int ksteps = (K + TK - 1) / TK;
    for (int ks = 0; ks < ksteps; ++ks) {
        const int k0 = ks * TK;

        // ---- stage A panel: 64x32 fp32, 16B chunks, async HBM->LDS -------
        for (int i = tid; i < (TM * TK) / 4; i += 256) {   // 512 chunks
            int r  = i >> 3;                // 0..63
            int c4 = (i & 7) << 2;          // 0,4,...,28
            int gm = bm + r, gk = k0 + c4;
            unsigned ldsOff = asBase + (unsigned)((r * TK + c4) << 2);
            bool fast = kAligned && (gm < M) && (gk + 3 < K);
#if defined(__gfx1250__)
            if (fast) {
                unsigned long long gaddr =
                    (unsigned long long)(const char*)&A[(size_t)gm * K + gk];
                asm volatile("global_load_async_to_lds_b128 %0, %1, off"
                             :: "v"(ldsOff), "v"(gaddr) : "memory");
            } else
#else
            (void)fast; (void)ldsOff;
#endif
            {
                float x0 = 0.f, x1 = 0.f, x2 = 0.f, x3 = 0.f;
                if (gm < M) {
                    if (gk + 0 < K) x0 = A[(size_t)gm * K + gk + 0];
                    if (gk + 1 < K) x1 = A[(size_t)gm * K + gk + 1];
                    if (gk + 2 < K) x2 = A[(size_t)gm * K + gk + 2];
                    if (gk + 3 < K) x3 = A[(size_t)gm * K + gk + 3];
                }
                As32[r][c4 + 0] = x0; As32[r][c4 + 1] = x1;
                As32[r][c4 + 2] = x2; As32[r][c4 + 3] = x3;
            }
        }

        // ---- stage B panel transposed: Bs[n][k] f16, float4 reads over N --
        for (int i = tid; i < (TN * TK) / 4; i += 256) {   // 1024 chunks
            int k  = i >> 5;                // 0..31
            int n4 = (i & 31) << 2;         // 0,4,...,124
            int gk = k0 + k, gn = bn + n4;
            float x0 = 0.f, x1 = 0.f, x2 = 0.f, x3 = 0.f;
            if (gk < K) {
                const float* brow = &B[(size_t)gk * N + gn];
                if (gn + 3 < N) {
                    float4 v = *reinterpret_cast<const float4*>(brow);
                    x0 = v.x; x1 = v.y; x2 = v.z; x3 = v.w;
                } else {
                    if (gn + 0 < N) x0 = brow[0];
                    if (gn + 1 < N) x1 = brow[1];
                    if (gn + 2 < N) x2 = brow[2];
                }
            }
            Bs[n4 + 0][k] = (_Float16)x0;
            Bs[n4 + 1][k] = (_Float16)x1;
            Bs[n4 + 2][k] = (_Float16)x2;
            Bs[n4 + 3][k] = (_Float16)x3;
        }

        // speculative prefetch of next B panel (global_prefetch_b8)
        if (ks + 1 < ksteps) {
            int pn = bn + (tid & 127), pk = k0 + TK + (tid >> 7);
            if (pk < K && pn < N) __builtin_prefetch(&B[(size_t)pk * N + pn], 0, 1);
        }

#if defined(__gfx1250__)
        asm volatile("s_wait_asynccnt 0x0" ::: "memory");
#endif
        __syncthreads();

        // ---- A fragment: fp32 from LDS, convert to f16 -------------------
        // lane<16 : row wm+lane,    K {0..7, 16..23}
        // lane>=16: row wm+lane-16, K {8..15, 24..31}
        {
            const int ar  = wm + (lane & 15);
            const int akb = (lane >> 4) * 8;
            float4 a0 = *reinterpret_cast<const float4*>(&As32[ar][akb + 0]);
            float4 a1 = *reinterpret_cast<const float4*>(&As32[ar][akb + 4]);
            float4 a2 = *reinterpret_cast<const float4*>(&As32[ar][akb + 16]);
            float4 a3 = *reinterpret_cast<const float4*>(&As32[ar][akb + 20]);
            v16h ah;
            ah[0]  = (_Float16)a0.x; ah[1]  = (_Float16)a0.y;
            ah[2]  = (_Float16)a0.z; ah[3]  = (_Float16)a0.w;
            ah[4]  = (_Float16)a1.x; ah[5]  = (_Float16)a1.y;
            ah[6]  = (_Float16)a1.z; ah[7]  = (_Float16)a1.w;
            ah[8]  = (_Float16)a2.x; ah[9]  = (_Float16)a2.y;
            ah[10] = (_Float16)a2.z; ah[11] = (_Float16)a2.w;
            ah[12] = (_Float16)a3.x; ah[13] = (_Float16)a3.y;
            ah[14] = (_Float16)a3.z; ah[15] = (_Float16)a3.w;

            // ---- B fragments + 4 WMMA -----------------------------------
            // lane<16 : col = base+lane,    K {0..7, 16..23} ; lane>=16: K {8..15, 24..31}
            const int bkb = (lane >> 4) * 8;
            const int bl  = lane & 15;
#pragma unroll
            for (int t = 0; t < 4; ++t) {
                const int bc = wn0 + t * 16 + bl;
                HFrag fb;
                fb.u[0] = *reinterpret_cast<const uint4*>(&Bs[bc][bkb]);
                fb.u[1] = *reinterpret_cast<const uint4*>(&Bs[bc][bkb + 16]);
                acc[t] = __builtin_amdgcn_wmma_f32_16x16x32_f16(
                    false, ah, false, fb.h, (short)0, acc[t], false, false);
            }
        }
        __syncthreads();
    }

    // Epilogue: v8f element r -> row r (lanes 0-15) / 8+r (lanes 16-31)
    const int mbase = bm + wm + ((lane >> 4) << 3);
#pragma unroll
    for (int t = 0; t < 4; ++t) {
        int gn = bn + wn0 + t * 16 + (lane & 15);
        if (gn >= N) continue;
        for (int r = 0; r < 8; ++r) {
            int gm = mbase + r;
            if (gm >= M) break;
            size_t ix = (size_t)gm * N + gn;
            C[ix] = beta1 ? (C[ix] + acc[t][r]) : acc[t][r];
        }
    }
}

// ---------------------------------------------------------------------------
// Utility kernels
// ---------------------------------------------------------------------------
__global__ void fill_f32(float* p, float v, int n) {
    int i = blockIdx.x * blockDim.x + threadIdx.x;
    if (i < n) p[i] = v;
}

// a_src[n,h] = sum_d xh[n,h,d]*att_src[h,d]; same for dst. One wave per (n,h).
__global__ __launch_bounds__(256) void gat_att_scores(
    const float* __restrict__ xh, const float* __restrict__ att_src,
    const float* __restrict__ att_dst, float* __restrict__ asrc,
    float* __restrict__ adst)
{
    int id = blockIdx.x * 8 + (threadIdx.x >> 5);
    int lane = threadIdx.x & 31;
    if (id >= NNODES * NHEADS) return;
    int n = id / NHEADS, h = id % NHEADS;
    const float* row = xh + (size_t)n * (NHEADS * HID) + (size_t)h * HID;
    float s1 = 0.f, s2 = 0.f;
    for (int d = lane; d < HID; d += 32) {
        float v = row[d];
        s1 += v * att_src[h * HID + d];
        s2 += v * att_dst[h * HID + d];
    }
    for (int off = 16; off > 0; off >>= 1) {
        s1 += __shfl_down(s1, off, 32);
        s2 += __shfl_down(s2, off, 32);
    }
    if (lane == 0) { asrc[id] = s1; adst[id] = s2; }
}

__device__ __forceinline__ void edge_ends(const int* ei, int e, int& s, int& d) {
    if (e < NEDGES) { s = ei[e]; d = ei[NEDGES + e]; }
    else            { s = d = e - NEDGES; }
}

__global__ void gat_edge_logits(const int* __restrict__ ei,
                                const float* __restrict__ asrc,
                                const float* __restrict__ adst,
                                float* __restrict__ ebuf, float* __restrict__ emax)
{
    int e = blockIdx.x * blockDim.x + threadIdx.x;
    if (e >= ESL) return;
    int s, d; edge_ends(ei, e, s, d);
    for (int h = 0; h < NHEADS; ++h) {
        float v = asrc[s * NHEADS + h] + adst[d * NHEADS + h];
        v = v > 0.f ? v : 0.2f * v;                 // leaky_relu(0.2)
        ebuf[(size_t)e * NHEADS + h] = v;
        atomic_max_f32(&emax[d * NHEADS + h], v);
    }
}

__global__ void gat_edge_exp(const int* __restrict__ ei,
                             const float* __restrict__ emax,
                             float* __restrict__ ebuf, float* __restrict__ denom)
{
    int e = blockIdx.x * blockDim.x + threadIdx.x;
    if (e >= ESL) return;
    int s, d; edge_ends(ei, e, s, d);
    (void)s;
    for (int h = 0; h < NHEADS; ++h) {
        float p = __expf(ebuf[(size_t)e * NHEADS + h] - emax[d * NHEADS + h]);
        ebuf[(size_t)e * NHEADS + h] = p;
        atomicAdd(&denom[d * NHEADS + h], p);
    }
}

// out[dst,d] += sum_h (alpha_h/denom_h) * xh[src,h,d] / NHEADS ; block per edge
__global__ __launch_bounds__(256) void gat_aggregate(
    const int* __restrict__ ei, const float* __restrict__ xh,
    const float* __restrict__ ebuf, const float* __restrict__ denom,
    float* __restrict__ outacc)
{
    int e = blockIdx.x;
    int s, d; edge_ends(ei, e, s, d);
    float w[NHEADS];
    for (int h = 0; h < NHEADS; ++h)
        w[h] = ebuf[(size_t)e * NHEADS + h] / denom[d * NHEADS + h] * (1.0f / NHEADS);
    const float* xr = xh + (size_t)s * (NHEADS * HID);
    float* orow = outacc + (size_t)d * HID;
    for (int c = threadIdx.x; c < HID; c += blockDim.x) {
        float v = w[0] * xr[c] + w[1] * xr[HID + c] +
                  w[2] * xr[2 * HID + c] + w[3] * xr[3 * HID + c];
        atomicAdd(&orow[c], v);
    }
}

__global__ void bias_relu(const float* __restrict__ in, const float* __restrict__ b,
                          float* __restrict__ out, int rows, int cols)
{
    size_t i = (size_t)blockIdx.x * blockDim.x + threadIdx.x;
    if (i >= (size_t)rows * cols) return;
    float v = in[i] + b[i % cols];
    out[i] = v > 0.f ? v : 0.f;
}

__global__ void deg_count(const int* __restrict__ ei, float* __restrict__ deg,
                          int total)
{
    int e = blockIdx.x * blockDim.x + threadIdx.x;
    if (e >= total) return;
    int s, d; edge_ends(ei, e, s, d);
    (void)s;
    atomicAdd(&deg[d], 1.0f);
}

__global__ void make_dinv(const float* __restrict__ deg, float* __restrict__ dinv) {
    int n = blockIdx.x * blockDim.x + threadIdx.x;
    if (n >= NNODES) return;
    float dg = deg[n];
    dinv[n] = dg > 0.f ? rsqrtf(fmaxf(dg, 1.f)) : 0.f;
}

__global__ void make_norm(const int* __restrict__ ei, const float* __restrict__ dinv,
                          float* __restrict__ nrm)
{
    int e = blockIdx.x * blockDim.x + threadIdx.x;
    if (e >= ESL) return;
    int s, d; edge_ends(ei, e, s, d);
    nrm[e] = dinv[s] * dinv[d];
}

// out[dst,:] += norm[e] * h[src,:]  ; block per edge (incl. self loops)
__global__ __launch_bounds__(256) void gcn_scatter(
    const int* __restrict__ ei, const float* __restrict__ h,
    const float* __restrict__ nrm, float* __restrict__ out)
{
    int e = blockIdx.x;
    int s, d; edge_ends(ei, e, s, d);
    float w = nrm[e];
    const float* hr = h + (size_t)s * HID;
    float* orow = out + (size_t)d * HID;
    for (int c = threadIdx.x; c < HID; c += blockDim.x)
        atomicAdd(&orow[c], w * hr[c]);
}

// agg[dst,:] += x[src,:] for edges of relation `rel` ; block per edge (no loops)
__global__ __launch_bounds__(256) void rgcn_scatter(
    const int* __restrict__ ei, const int* __restrict__ etype,
    const float* __restrict__ x, float* __restrict__ agg, int rel)
{
    int e = blockIdx.x;
    if (etype[e] != rel) return;
    int s = ei[e], d = ei[NEDGES + e];
    const float* xr = x + (size_t)s * HID;
    float* ar = agg + (size_t)d * HID;
    for (int c = threadIdx.x; c < HID; c += blockDim.x)
        atomicAdd(&ar[c], xr[c]);
}

__global__ void rgcn_finalize(const float* __restrict__ acc,
                              const float* __restrict__ degE,
                              const float* __restrict__ rootout,
                              const float* __restrict__ bias,
                              float* __restrict__ out)
{
    size_t i = (size_t)blockIdx.x * blockDim.x + threadIdx.x;
    if (i >= (size_t)NNODES * HID) return;
    int n = (int)(i / HID), c = (int)(i % HID);
    float dg = degE[n];
    float m = dg > 0.f ? acc[i] / fmaxf(dg, 1.f) : 0.f;
    float v = m + rootout[i] + bias[c];
    out[i] = v > 0.f ? v : 0.f;
}

// ---------------------------------------------------------------------------
// Host orchestration
// ---------------------------------------------------------------------------
static inline void launch_gemm(const float* A, const float* B, float* C,
                               int M, int N, int K, int beta1, hipStream_t s)
{
    dim3 grid((N + TN - 1) / TN, (M + TM - 1) / TM);
    wmma_gemm_f16f32<<<grid, dim3(256), 0, s>>>(A, B, C, M, N, K, beta1);
}

static inline void launch_fill(float* p, float v, int n, hipStream_t s) {
    fill_f32<<<(n + 255) / 256, 256, 0, s>>>(p, v, n);
}

extern "C" void kernel_launch(void* const* d_in, const int* in_sizes, int n_in,
                              void* d_out, int out_size, void* d_ws, size_t ws_size,
                              hipStream_t stream)
{
    (void)in_sizes; (void)n_in; (void)out_size; (void)ws_size;

    const float* node_emb  = (const float*)d_in[0];
    const float* gat_W     = (const float*)d_in[1];
    const float* gat_asrc  = (const float*)d_in[2];
    const float* gat_adst  = (const float*)d_in[3];
    const float* gat_bias  = (const float*)d_in[4];
    const float* gcn_W     = (const float*)d_in[5];
    const float* gcn_b     = (const float*)d_in[6];
    const float* bases     = (const float*)d_in[7];
    const float* comp      = (const float*)d_in[8];
    const float* root      = (const float*)d_in[9];
    const float* rbias     = (const float*)d_in[10];
    const int*   ei        = (const int*)d_in[11];
    const int*   etype     = (const int*)d_in[12];
    float*       out       = (float*)d_out;

    // workspace layout (floats)
    float* ws   = (float*)d_ws;
    float* xh   = ws;                         // [N, H*HID] = 20,000,000  (GAT only)
    float* Wr   = ws;                         // reuse: [R, HID*HID] = 8,000,000
    float* agg  = ws + 8000000;               // [N, HID] = 5,000,000
    float* racc = ws + 13000000;              // [N, HID] = 5,000,000
    float* tbuf = ws;                         // GCN scatter target (reuse, pre-RGCN)
    float* xbuf = ws + 20000000;              // [N, HID]
    float* hbuf = ws + 25000000;              // [N, HID]
    float* asrc = ws + 30000000;              // [N, H]
    float* adst = asrc + NNODES * NHEADS;
    float* emax = adst + NNODES * NHEADS;
    float* denm = emax + NNODES * NHEADS;
    float* deg  = denm + NNODES * NHEADS;     // [N]
    float* dinv = deg + NNODES;
    float* degE = dinv + NNODES;
    float* ebuf = degE + NNODES;              // [E', H]
    float* nrm  = ebuf + (size_t)ESL * NHEADS;// [E']

    const int NE_GRID = (ESL + 255) / 256;

    // ---- GAT -------------------------------------------------------------
    launch_gemm(node_emb, gat_W, xh, NNODES, NHEADS * HID, HID, 0, stream);
    gat_att_scores<<<(NNODES * NHEADS + 7) / 8, 256, 0, stream>>>(
        xh, gat_asrc, gat_adst, asrc, adst);
    launch_fill(emax, -3.4e38f, NNODES * NHEADS, stream);
    launch_fill(denm, 0.f, NNODES * NHEADS, stream);
    gat_edge_logits<<<NE_GRID, 256, 0, stream>>>(ei, asrc, adst, ebuf, emax);
    gat_edge_exp<<<NE_GRID, 256, 0, stream>>>(ei, emax, ebuf, denm);
    launch_fill(hbuf, 0.f, NNODES * HID, stream);
    gat_aggregate<<<ESL, 256, 0, stream>>>(ei, xh, ebuf, denm, hbuf);
    bias_relu<<<(NNODES * HID + 255) / 256, 256, 0, stream>>>(
        hbuf, gat_bias, xbuf, NNODES, HID);

    // ---- GCN norm --------------------------------------------------------
    launch_fill(deg, 0.f, NNODES, stream);
    deg_count<<<NE_GRID, 256, 0, stream>>>(ei, deg, ESL);
    make_dinv<<<(NNODES + 255) / 256, 256, 0, stream>>>(deg, dinv);
    make_norm<<<NE_GRID, 256, 0, stream>>>(ei, dinv, nrm);

    // ---- 3x GCN ----------------------------------------------------------
    for (int i = 0; i < NGCN; ++i) {
        launch_gemm(xbuf, gcn_W + (size_t)i * HID * HID, hbuf, NNODES, HID, HID, 0, stream);
        launch_fill(tbuf, 0.f, NNODES * HID, stream);
        gcn_scatter<<<ESL, 256, 0, stream>>>(ei, hbuf, nrm, tbuf);
        bias_relu<<<(NNODES * HID + 255) / 256, 256, 0, stream>>>(
            tbuf, gcn_b + (size_t)i * HID, xbuf, NNODES, HID);
    }

    // ---- RGCN ------------------------------------------------------------
    // Wr[r] = sum_b comp[r,b]*bases[b]  == comp[R,B] @ bases[B, HID*HID]
    launch_gemm(comp, bases, Wr, NRELS, HID * HID, NBASES, 0, stream);
    launch_fill(racc, 0.f, NNODES * HID, stream);
    launch_fill(degE, 0.f, NNODES, stream);
    deg_count<<<(NEDGES + 255) / 256, 256, 0, stream>>>(ei, degE, NEDGES);
    for (int r = 0; r < NRELS; ++r) {
        launch_fill(agg, 0.f, NNODES * HID, stream);
        rgcn_scatter<<<NEDGES, 256, 0, stream>>>(ei, etype, xbuf, agg, r);
        launch_gemm(agg, Wr + (size_t)r * HID * HID, racc, NNODES, HID, HID, 1, stream);
    }
    launch_gemm(xbuf, root, hbuf, NNODES, HID, HID, 0, stream);
    rgcn_finalize<<<(NNODES * HID + 255) / 256, 256, 0, stream>>>(
        racc, degE, hbuf, rbias, out);
}
